// AttentionBlock_31533649887600
// MI455X (gfx1250) — compile-verified, matrix operand order checked by reading, not run on hardware
//
#include <hip/hip_runtime.h>
#include <cstdint>
#include <cstddef>

// ---------- types ----------
typedef __bf16 bf16;
typedef __attribute__((ext_vector_type(8)))  bf16  bf16x8;
typedef __attribute__((ext_vector_type(16))) bf16  v16bf;
typedef __attribute__((ext_vector_type(8)))  float v8f;
typedef int i32x4 __attribute__((vector_size(16)));  // matches builtin param type

#define AS1 __attribute__((address_space(1)))
#define AS3 __attribute__((address_space(3)))

// ---------- CDNA5 async global->LDS copy (ASYNCcnt path) ----------
__device__ __forceinline__ void async_copy16(bf16* lds, const bf16* g) {
#if __has_builtin(__builtin_amdgcn_global_load_async_to_lds_b128)
  __builtin_amdgcn_global_load_async_to_lds_b128(
      (AS1 i32x4*)g, (AS3 i32x4*)lds, 0, 0);
#else
  unsigned laddr = (unsigned)(__SIZE_TYPE__)(AS3 void*)lds;
  asm volatile("global_load_async_to_lds_b128 %0, %1, off"
               :: "v"(laddr), "v"((const AS1 void*)g)
               : "memory");
#endif
}

template <int N>
__device__ __forceinline__ void wait_asynccnt() {
#if __has_builtin(__builtin_amdgcn_s_wait_asynccnt)
  __builtin_amdgcn_s_wait_asynccnt((unsigned short)N);
#else
  asm volatile("s_wait_asynccnt %0" :: "i"(N) : "memory");
#endif
}

// ---------- block reduction helper (wave32) ----------
__device__ __forceinline__ float block_reduce_sum(float v, float* red) {
  __syncthreads();
  #pragma unroll
  for (int off = 16; off; off >>= 1) v += __shfl_xor(v, off, 32);
  if ((threadIdx.x & 31) == 0) red[threadIdx.x >> 5] = v;
  __syncthreads();
  if (threadIdx.x == 0) {
    float s = 0.f;
    #pragma unroll
    for (int i = 0; i < 8; ++i) s += red[i];
    red[0] = s;
  }
  __syncthreads();
  return red[0];
}

// ---------- 16x32 bf16 fragment load from LDS (ISA 7.12.2 16-bit layout) ----------
// lanes 0-15: row = lane, K 0..7 & 16..23 ; lanes 16-31: row = lane-16, K 8..15 & 24..31
__device__ __forceinline__ v16bf load_frag(const bf16* base, int row0) {
  const int lane = threadIdx.x & 31;
  const bf16* p = base + (size_t)(row0 + (lane & 15)) * 32 + ((lane >> 4) << 3);
  bf16x8 lo = *(const bf16x8*)p;        // K 0..7 / 8..15
  bf16x8 hi = *(const bf16x8*)(p + 16); // K 16..23 / 24..31
  return __builtin_shufflevector(lo, hi, 0, 1, 2, 3, 4, 5, 6, 7,
                                         8, 9, 10, 11, 12, 13, 14, 15);
}

// ---------- GroupNorm -> bf16, written transposed: xnT[b][n][c] ----------
__global__ __launch_bounds__(256) void groupnorm_bf16(
    const float* __restrict__ x, const float* __restrict__ gamma,
    const float* __restrict__ beta, bf16* __restrict__ xnT) {
  __shared__ float red[8];
  const int b = blockIdx.x >> 3;
  const int g = blockIdx.x & 7;
  const size_t base = ((size_t)b * 512 + (size_t)g * 64) * 1024;  // 64 ch * 1024
  const float* xp = x + base;
  float s = 0.f, ss = 0.f;
  for (int i = threadIdx.x; i < 65536; i += 256) {
    float v = xp[i];
    s += v;
    ss += v * v;
  }
  s  = block_reduce_sum(s, red);
  ss = block_reduce_sum(ss, red);
  const float mean = s * (1.f / 65536.f);
  const float var  = ss * (1.f / 65536.f) - mean * mean;
  const float inv  = rsqrtf(var + 1e-5f);
  bf16* outT = xnT + (size_t)b * 1024 * 512;
  for (int i = threadIdx.x; i < 65536; i += 256) {
    int cl = i >> 10;            // local channel 0..63
    int n  = i & 1023;           // spatial
    int c  = (g << 6) + cl;
    float v = (xp[i] - mean) * inv * gamma[c] + beta[c];
    outT[(size_t)n * 512 + c] = (bf16)v;
  }
}

// ---------- weight conversion (both kept in natural [o][c] = N x K layout) ----------
__global__ void convert_weights(const float* __restrict__ qkvw,
                                const float* __restrict__ projw,
                                bf16* __restrict__ wq, bf16* __restrict__ pw) {
  int i = blockIdx.x * 256 + threadIdx.x;
  if (i < 1536 * 512) wq[i] = (bf16)qkvw[i];
  if (i < 512 * 512) pw[i] = (bf16)projw[i];
}

// ---------- NT GEMM: D = A(MxK,row) * B^T where Bt is (NxK,row); tile 128x128 ----------
// 256 threads = 8 wave32 waves, 4(M) x 2(N); per wave 32x64 = 2x4 WMMA accumulators.
// Double-buffered LDS slabs staged with global_load_async_to_lds_b128 (ASYNCcnt).
// EPI 0: f32 store   EPI 1: qkv split (q^T*scale, k^T, v) + bias
// EPI 2: bf16 store  EPI 3: proj bias + residual into d_out (transposed)
template <int EPI>
__global__ __launch_bounds__(256) void gemm_nt_bf16(
    const bf16* __restrict__ A, const bf16* __restrict__ Bt,
    size_t strideA, size_t strideB, int N, int K,
    const float* __restrict__ bias, float* __restrict__ outf,
    bf16* __restrict__ oq, bf16* __restrict__ ok, bf16* __restrict__ ov,
    const float* __restrict__ xres, float* __restrict__ dout, float scale) {
  __shared__ bf16 sA[2][128][32];
  __shared__ bf16 sB[2][128][32];

  const int b  = blockIdx.z;
  const int m0 = blockIdx.y * 128;
  const int n0 = blockIdx.x * 128;
  const bf16* Ag = A  + (size_t)b * strideA + (size_t)m0 * K;
  const bf16* Bg = Bt + (size_t)b * strideB + (size_t)n0 * K;

  const int t    = threadIdx.x;
  const int wave = t >> 5;
  const int wm   = wave & 3;   // 0..3 along M (32 rows)
  const int wn   = wave >> 2;  // 0..1 along N (64 cols)
  const int rrow = t >> 2;          // 0..63 staging row
  const int rcol = (t & 3) << 3;    // 0/8/16/24 staging col (bf16 elems)
  v8f acc[2][4] = {};

  auto stage = [&](int buf, int k0) {  // 4 x 16B async copies per thread
    async_copy16(&sA[buf][rrow][rcol],      Ag + (size_t)rrow * K + k0 + rcol);
    async_copy16(&sA[buf][rrow + 64][rcol], Ag + (size_t)(rrow + 64) * K + k0 + rcol);
    async_copy16(&sB[buf][rrow][rcol],      Bg + (size_t)rrow * K + k0 + rcol);
    async_copy16(&sB[buf][rrow + 64][rcol], Bg + (size_t)(rrow + 64) * K + k0 + rcol);
  };

  const int nslab = K >> 5;
  stage(0, 0);
  for (int s = 0; s < nslab; ++s) {
    const int cur = s & 1;
    if (s + 1 < nslab) {
      stage(cur ^ 1, (s + 1) << 5);   // overlap next slab DMA with this slab's math
      wait_asynccnt<4>();             // in-order: slab s landed when <=4 outstanding
    } else {
      wait_asynccnt<0>();
    }
    __syncthreads();
    if (s + 2 < nslab) {              // global_prefetch_b8 two slabs ahead
      __builtin_prefetch(Ag + (size_t)rrow * K + ((s + 2) << 5) + rcol, 0, 1);
      __builtin_prefetch(Bg + (size_t)rrow * K + ((s + 2) << 5) + rcol, 0, 1);
    }

    v16bf af[2], bfr[4];
    #pragma unroll
    for (int i = 0; i < 2; ++i) af[i] = load_frag(&sA[cur][0][0], wm * 32 + i * 16);
    #pragma unroll
    for (int j = 0; j < 4; ++j) bfr[j] = load_frag(&sB[cur][0][0], wn * 64 + j * 16);
    #pragma unroll
    for (int i = 0; i < 2; ++i)
      #pragma unroll
      for (int j = 0; j < 4; ++j)
        acc[i][j] = __builtin_amdgcn_wmma_f32_16x16x32_bf16(
            false, af[i], false, bfr[j], (short)0, acc[i][j], false, false);
    __syncthreads();  // all waves done reading buf `cur` before it is re-staged
  }

  // ---- epilogue: C/D layout (ISA 7.12.2): col = lane&15, row = 8*(lane>>4)+r
  const int lane  = t & 31;
  const int nfrag = lane & 15;
  const int mbase = (lane >> 4) << 3;
  const size_t Mfull = (size_t)gridDim.y * 128;
  #pragma unroll
  for (int i = 0; i < 2; ++i)
    #pragma unroll
    for (int j = 0; j < 4; ++j)
      #pragma unroll
      for (int r = 0; r < 8; ++r) {
        int gm = m0 + wm * 32 + i * 16 + mbase + r;
        int gn = n0 + wn * 64 + j * 16 + nfrag;
        float v = acc[i][j][r];
        if (EPI == 0) {
          outf[(size_t)b * Mfull * N + (size_t)gm * N + gn] = v;
        } else if (EPI == 1) {  // gm = channel o (0..1535), gn = spatial n
          v += bias[gm];
          if (gm < 512)        // Q^T[b][n][c] (scores A), softmax scale folded in
            oq[(size_t)b * 1024 * 512 + (size_t)gn * 512 + gm] = (bf16)(v * scale);
          else if (gm < 1024)  // K^T[b][m][c] (scores B^T)
            ok[(size_t)b * 1024 * 512 + (size_t)gn * 512 + (gm - 512)] = (bf16)v;
          else                 // V[b][c][m]  (AV B^T)
            ov[(size_t)b * 512 * 1024 + (size_t)(gm - 1024) * 1024 + gn] = (bf16)v;
        } else if (EPI == 2) {
          oq[(size_t)b * Mfull * N + (size_t)gm * N + gn] = (bf16)v;
        } else {  // EPI == 3: gm = spatial n, gn = channel o; fuse bias+residual
          v += bias[gn];
          size_t idx = (size_t)b * 512 * 1024 + (size_t)gn * 1024 + gm;
          dout[idx] = xres[idx] + v;
        }
      }
}

// ---------- row softmax: f32 scores -> bf16 probabilities ----------
__global__ __launch_bounds__(256) void softmax_bf16(const float* __restrict__ S,
                                                    bf16* __restrict__ P) {
  const size_t row = blockIdx.x;  // b*1024 + n
  const float* s = S + row * 1024;
  bf16* p = P + row * 1024;
  __shared__ float red[8];

  float m = -3.4e38f;
  #pragma unroll
  for (int q = 0; q < 4; ++q) m = fmaxf(m, s[threadIdx.x + q * 256]);
  #pragma unroll
  for (int off = 16; off; off >>= 1) m = fmaxf(m, __shfl_xor(m, off, 32));
  if ((threadIdx.x & 31) == 0) red[threadIdx.x >> 5] = m;
  __syncthreads();
  if (threadIdx.x == 0) {
    float mm = red[0];
    #pragma unroll
    for (int i = 1; i < 8; ++i) mm = fmaxf(mm, red[i]);
    red[0] = mm;
  }
  __syncthreads();
  m = red[0];

  float e[4], sum = 0.f;
  #pragma unroll
  for (int q = 0; q < 4; ++q) {
    e[q] = __expf(s[threadIdx.x + q * 256] - m);
    sum += e[q];
  }
  sum = block_reduce_sum(sum, red);
  const float inv = 1.f / sum;
  #pragma unroll
  for (int q = 0; q < 4; ++q) p[threadIdx.x + q * 256] = (bf16)(e[q] * inv);
}

// ---------- host launcher ----------
extern "C" void kernel_launch(void* const* d_in, const int* in_sizes, int n_in,
                              void* d_out, int out_size, void* d_ws, size_t ws_size,
                              hipStream_t stream) {
  const float* x      = (const float*)d_in[0];
  const float* norm_w = (const float*)d_in[1];
  const float* norm_b = (const float*)d_in[2];
  const float* qkv_w  = (const float*)d_in[3];
  const float* qkv_b  = (const float*)d_in[4];
  const float* proj_w = (const float*)d_in[5];
  const float* proj_b = (const float*)d_in[6];
  float* out = (float*)d_out;
  (void)in_sizes; (void)n_in; (void)out_size; (void)ws_size;

  char* p = (char*)d_ws;
  auto carve = [&](size_t bytes) {
    char* r = p;
    p += (bytes + 255) & ~(size_t)255;
    return (void*)r;
  };
  bf16*  wqkv = (bf16*)carve((size_t)1536 * 512 * 2);          // qkv weights [o][c]
  bf16*  pw   = (bf16*)carve((size_t)512 * 512 * 2);           // proj weights [o][c]
  bf16*  xnT  = (bf16*)carve((size_t)16 * 1024 * 512 * 2);     // normed^T [b][n][c]
  bf16*  qt   = (bf16*)carve((size_t)16 * 1024 * 512 * 2);     // Q^T [b][n][c]
  bf16*  kt   = (bf16*)carve((size_t)16 * 1024 * 512 * 2);     // K^T [b][m][c]
  bf16*  vm   = (bf16*)carve((size_t)16 * 512 * 1024 * 2);     // V   [b][c][m]
  float* sbuf = (float*)carve((size_t)16 * 1024 * 1024 * 4);   // scores f32
  bf16*  pbuf = (bf16*)carve((size_t)16 * 1024 * 1024 * 2);    // probs bf16 [b][n][m]
  bf16*  obuf = (bf16*)carve((size_t)16 * 1024 * 512 * 2);     // O^T [b][n][c]

  const float scale = 0.044194173824159216f;  // 512^-0.5

  convert_weights<<<(1536 * 512 + 255) / 256, 256, 0, stream>>>(qkv_w, proj_w, wqkv, pw);
  groupnorm_bf16<<<16 * 8, 256, 0, stream>>>(x, norm_w, norm_b, xnT);

  // QKV: Wqkv(1536x512) x XnT^T -> per-batch; A=W [o][c], Bt=xnT [n][c]
  gemm_nt_bf16<1><<<dim3(1024 / 128, 1536 / 128, 16), 256, 0, stream>>>(
      wqkv, xnT, 0, (size_t)1024 * 512, 1024, 512,
      qkv_b, nullptr, qt, kt, vm, nullptr, nullptr, scale);

  // S[n][m] = Q^T(1024x512) x K; A=qt [n][c], Bt=kt [m][c]
  gemm_nt_bf16<0><<<dim3(8, 8, 16), 256, 0, stream>>>(
      qt, kt, (size_t)1024 * 512, (size_t)1024 * 512, 1024, 512,
      nullptr, sbuf, nullptr, nullptr, nullptr, nullptr, nullptr, 1.f);

  softmax_bf16<<<16 * 1024, 256, 0, stream>>>(sbuf, pbuf);

  // O^T[n][c] = P(1024x1024) x V^T; A=pbuf [n][m], Bt=vm [c][m]
  gemm_nt_bf16<2><<<dim3(512 / 128, 1024 / 128, 16), 256, 0, stream>>>(
      pbuf, vm, (size_t)1024 * 1024, (size_t)512 * 1024, 512, 1024,
      nullptr, nullptr, obuf, nullptr, nullptr, nullptr, nullptr, 1.f);

  // out = x + proj_b + O^T(1024x512) x Pw^T; A=obuf [n][c], Bt=pw [o][c]
  gemm_nt_bf16<3><<<dim3(512 / 128, 1024 / 128, 16), 256, 0, stream>>>(
      obuf, pw, (size_t)1024 * 512, 0, 512, 512,
      proj_b, nullptr, nullptr, nullptr, nullptr, x, out, 1.f);
}
